// GGMLLinear_20512763805980
// MI455X (gfx1250) — matrix operand order repaired and checked by reading, not compiled, not run on hardware
//
#include <hip/hip_runtime.h>

typedef __attribute__((ext_vector_type(16))) _Float16 v16h;
typedef __attribute__((ext_vector_type(8)))  float    v8f;

#define IN_F       4096
#define OUT_F      32000
#define QK         32
#define NBLK       (IN_F / QK)      // 128 blocks per row
#define BLK_BYTES  34
#define ROW_BYTES  (NBLK * BLK_BYTES)   // 4352
#define WAVES_PER_BLOCK 8
#define ROWS_PER_WAVE   16

// One wave32 computes 16 output rows. Per Q8_0 block (K=32) it builds the
// 16x32 f16 A-tile (dequantized in-register: a = scale * q, q exact in f16),
// broadcasts x[k..k+31] (f16, from LDS) into every column of B, and issues
// one v_wmma_f32_16x16x32_f16. f32 accumulation across all 128 blocks.
__global__ __launch_bounds__(256) void q8_gemv_wmma(
    const float* __restrict__ x,
    const unsigned char* __restrict__ qw,
    const float* __restrict__ bias,
    float* __restrict__ out)
{
    __shared__ _Float16 xs[IN_F];           // 8 KB of the 320 KB/WGP LDS
    const int tid = threadIdx.x;
    for (int i = tid; i < IN_F; i += 256)
        xs[i] = (_Float16)x[i];
    __syncthreads();

    const int lane = tid & 31;
    const int wave = tid >> 5;
    const int hi   = (lane >> 4) & 1;       // lane group: 0 -> K lo half, 1 -> K hi half
    const int rowGroup = blockIdx.x * WAVES_PER_BLOCK + wave;
    const int rowBase  = rowGroup * ROWS_PER_WAVE;
    const int myRow    = rowBase + (lane & 15);

    const unsigned char* rowPtr = qw + (size_t)myRow * ROW_BYTES;

    v8f acc = {};
    #pragma unroll 2
    for (int b = 0; b < NBLK; ++b) {
        const unsigned char* blk = rowPtr + b * BLK_BYTES;

        // fp16 scale (block start is always 2-byte aligned)
        union { unsigned short u; _Float16 h; } cv;
        cv.u = *(const unsigned short*)blk;
        const _Float16 d = cv.h;

        // A-tile layout (16-bit A, 16x32, wave32):
        //   lanes 0-15 : row=lane,    a[0..15]=K 0..15
        //   lanes 16-31: row=lane-16, a[0..15]=K 16..31
        const unsigned short* qp = (const unsigned short*)(blk + 2 + (hi << 4));

        v16h a;
        #pragma unroll
        for (int t = 0; t < 4; ++t) {
            const unsigned short u0 = qp[t];       // quant bytes +0..+7
            const unsigned short u1 = qp[t + 4];   // quant bytes +8..+15
            a[2*t    ]     = d * (_Float16)(signed char)(u0 & 0xFF);
            a[2*t + 1]     = d * (_Float16)(signed char)(u0 >> 8);
            a[8 + 2*t    ] = d * (_Float16)(signed char)(u1 & 0xFF);
            a[8 + 2*t + 1] = d * (_Float16)(signed char)(u1 >> 8);
        }

        // B-tile: every column = x chunk. Layout (16-bit B, 32x16):
        //   lanes 0-15 : b[j] = x[b*32 + j],      j=0..15
        //   lanes 16-31: b[j] = x[b*32 + 16 + j]
        // 32-byte aligned LDS read, broadcast across lanes (conflict-free).
        const v16h bv = *(const v16h*)(&xs[b * QK + (hi << 4)]);

        acc = __builtin_amdgcn_wmma_f32_16x16x32_f16(
            /*neg_a=*/false, a, /*neg_b=*/false, bv,
            /*c_mod=*/(short)0, acc, /*reuse_a=*/false, /*reuse_b=*/false);
    }

    // D layout: lanes 0-15, VGPR r = row r; lanes 16-31, VGPR r = row 8+r.
    // All 16 columns identical -> lane 0 writes rows 0-7, lane 16 rows 8-15.
    if ((lane & 15) == 0) {
        const int m0 = rowBase + (hi ? 8 : 0);
        #pragma unroll
        for (int r = 0; r < 8; ++r)
            out[m0 + r] = acc[r] + bias[m0 + r];
    }
}

extern "C" void kernel_launch(void* const* d_in, const int* in_sizes, int n_in,
                              void* d_out, int out_size, void* d_ws, size_t ws_size,
                              hipStream_t stream) {
    (void)in_sizes; (void)n_in; (void)d_ws; (void)ws_size; (void)out_size;
    const float*         x    = (const float*)d_in[0];
    const unsigned char* qwt  = (const unsigned char*)d_in[1];
    const float*         bias = (const float*)d_in[2];
    float*               out  = (float*)d_out;

    const int rowGroups = OUT_F / ROWS_PER_WAVE;        // 2000 waves of work
    const int blocks    = rowGroups / WAVES_PER_BLOCK;  // 250 blocks x 256 thr
    q8_gemv_wmma<<<blocks, 256, 0, stream>>>(x, qwt, bias, out);
}
